// Attention_35072702939482
// MI455X (gfx1250) — compile-verified, matrix operand order checked by reading, not run on hardware
//
#include <hip/hip_runtime.h>

typedef __attribute__((ext_vector_type(8)))  float          v8f;
typedef __attribute__((ext_vector_type(16))) __bf16         v16bf;
typedef __attribute__((ext_vector_type(8)))  unsigned int   v8u;
typedef __attribute__((ext_vector_type(2)))  unsigned int   u32x2;
typedef __attribute__((ext_vector_type(8)))  unsigned short u16x8;
typedef __attribute__((ext_vector_type(4)))  float          f32x4;
typedef __attribute__((ext_vector_type(4)))  int            i32x4;

#define S_DIM 1024
#define B_DIM 64
#define C_DIM 1024
#define H_DIM 1024
#define MID_D 1024

#if __has_builtin(__builtin_amdgcn_global_load_async_to_lds_b128)
#define HAVE_ASYNC_LDS 1
#else
#define HAVE_ASYNC_LDS 0
#endif

// Pack two fp32 into packed bf16x2 (round-half-up + v_perm_b32): lo -> [15:0].
__device__ __forceinline__ unsigned pack2bf(float lo, float hi) {
  unsigned a = __float_as_uint(lo) + 0x8000u;
  unsigned b = __float_as_uint(hi) + 0x8000u;
  // D.byte0=S1.b2, b1=S1.b3, b2=S0.b2, b3=S0.b3
  return __builtin_amdgcn_perm(b, a, 0x07060302u);
}

__device__ __forceinline__ v8f wmma_bf16(v16bf a, v16bf b, v8f c) {
  return __builtin_amdgcn_wmma_f32_16x16x32_bf16(
      /*neg_a=*/false, a, /*neg_b=*/false, b,
      /*c_mod=*/(short)0, c, /*reuse_a=*/false, /*reuse_b=*/false);
}

__device__ __forceinline__ void async_cp16(const void* g, void* l) {
#if HAVE_ASYNC_LDS
  __builtin_amdgcn_global_load_async_to_lds_b128(
      (__attribute__((address_space(1))) i32x4*)(g),
      (__attribute__((address_space(3))) i32x4*)(l), 0, 0);
#endif
}

__device__ __forceinline__ void wait_async0() {
#if __has_builtin(__builtin_amdgcn_s_wait_asynccnt)
  __builtin_amdgcn_s_wait_asynccnt(0);
#else
  asm volatile("s_wait_asynccnt 0" ::: "memory");
#endif
}

// Build a lane's 16x32 bf16 fragment from two 8-float global runs
// (ISA 16-bit layout: lane holds K = g*8 + [0..7] and 16 + g*8 + [0..7]).
__device__ __forceinline__ v16bf frag_from_global(const float* p0, const float* p1) {
  f32x4 a0 = *(const f32x4*)p0;
  f32x4 a1 = *(const f32x4*)(p0 + 4);
  f32x4 b0 = *(const f32x4*)p1;
  f32x4 b1 = *(const f32x4*)(p1 + 4);
  v8u u;
  u[0] = pack2bf(a0[0], a0[1]); u[1] = pack2bf(a0[2], a0[3]);
  u[2] = pack2bf(a1[0], a1[1]); u[3] = pack2bf(a1[2], a1[3]);
  u[4] = pack2bf(b0[0], b0[1]); u[5] = pack2bf(b0[2], b0[3]);
  u[6] = pack2bf(b1[0], b1[1]); u[7] = pack2bf(b1[2], b1[3]);
  return __builtin_bit_cast(v16bf, u);
}

__device__ __forceinline__ v16bf frag_from_lds(const unsigned short* p0,
                                               const unsigned short* p1) {
  u16x8 a = *(const u16x8*)p0;
  u16x8 b = *(const u16x8*)p1;
  v8u u;
  unsigned* w        = (unsigned*)&u;
  const unsigned* pa = (const unsigned*)&a;
  const unsigned* pb = (const unsigned*)&b;
#pragma unroll
  for (int i = 0; i < 4; ++i) { w[i] = pa[i]; w[i + 4] = pb[i]; }
  return __builtin_bit_cast(v16bf, u);
}

// One-shot fp32 -> bf16 conversion of Wc (1024x1024) into workspace.
__global__ __launch_bounds__(256) void cvt_bf16_kernel(const float* __restrict__ src,
                                                       unsigned short* __restrict__ dst) {
  const int i = blockIdx.x * 256 + threadIdx.x;  // one f32x4 per thread
  f32x4 v = ((const f32x4*)src)[i];
  u32x2 p;
  p[0] = pack2bf(v[0], v[1]);
  p[1] = pack2bf(v[2], v[3]);
  *(u32x2*)&dst[(size_t)i * 4] = p;
}

// out[64 x 1024] = A[64 x 1024] @ B[1024 x 1024]^T  (both row-major, K contiguous)
__global__ __launch_bounds__(256) void gemm64nt_kernel(const float* __restrict__ A,
                                                       const float* __restrict__ Bm,
                                                       float* __restrict__ out) {
  const int tid  = threadIdx.x;
  const int lane = tid & 31, w = tid >> 5;
  const int g = lane >> 4, lm = lane & 15;
  const int n0 = (blockIdx.x * 8 + w) * 16;
  const float* brow = Bm + (size_t)(n0 + lm) * 1024;
#pragma unroll 1
  for (int mt = 0; mt < 4; ++mt) {
    const float* arow = A + (size_t)(mt * 16 + lm) * 1024;
    v8f acc = {};
#pragma unroll 4
    for (int kb = 0; kb < 32; ++kb) {
      const int ka = kb * 32 + g * 8;
      v16bf af = frag_from_global(arow + ka, arow + ka + 16);
      v16bf bf = frag_from_global(brow + ka, brow + ka + 16);
      acc = wmma_bf16(af, bf, acc);
    }
#pragma unroll
    for (int v = 0; v < 8; ++v)
      out[(size_t)(mt * 16 + v + 8 * g) * 1024 + n0 + lm] = acc[v];
  }
}

#define A_STRIDE 1032  // 1024 + 8 ushorts pad (16B) vs LDS bank conflicts

// Fused: scores[r] = sum_m wm[m] * tanh( (ctx @ Wc^T)[r,m] + hid_[r%64, m] )
// Block: 256 thr = 8 waves, 128 rows of (S*B); A staged bf16 in LDS (264KB);
// B tiles (pre-converted bf16 Wc) staged via async global->LDS copies.
__global__ __launch_bounds__(256) void scores_kernel(const float* __restrict__ ctx,
                                                     const unsigned short* __restrict__ Wcb,
                                                     const float* __restrict__ hidp,
                                                     const float* __restrict__ wm,
                                                     float* __restrict__ scores) {
  __shared__ unsigned short Alds[128 * A_STRIDE];
  __shared__ unsigned short Blds[16 * A_STRIDE];
  const int tid  = threadIdx.x;
  const int lane = tid & 31, w = tid >> 5;
  const int g = lane >> 4, lm = lane & 15;
  const int r0 = blockIdx.x * 128;

  // Phase 1: stage 128 contiguous ctx rows (fp32 -> bf16) into LDS
  {
    const f32x4* src = (const f32x4*)(ctx + (size_t)r0 * 1024);
    for (int i = tid; i < (128 * 1024 / 4); i += 256) {
      f32x4 v = src[i];
      const int e = i << 2;
      const int row = e >> 10, col = e & 1023;
      u32x2 p;
      p[0] = pack2bf(v[0], v[1]);
      p[1] = pack2bf(v[2], v[3]);
      *(u32x2*)&Alds[row * A_STRIDE + col] = p;
    }
  }

  const int mrow = w * 16;
  const unsigned short* arow = &Alds[(mrow + lm) * A_STRIDE];
  const unsigned short* brow = &Blds[lm * A_STRIDE];
  float sAcc[8];
#pragma unroll
  for (int v = 0; v < 8; ++v) sAcc[v] = 0.f;

  for (int nt = 0; nt < 64; ++nt) {
    const int n0 = nt * 16;
    __syncthreads();
    {  // stage bf16 Wc rows n0..n0+15 into LDS (pure copy, 32KB)
      const unsigned short* wsrc = Wcb + (size_t)n0 * 1024;
#if HAVE_ASYNC_LDS
#pragma unroll
      for (int it = 0; it < 8; ++it) {
        const int c   = tid + it * 256;     // b128 chunk index
        const int e   = c << 3;             // element index
        const int row = e >> 10, col = e & 1023;
        async_cp16(wsrc + e, &Blds[row * A_STRIDE + col]);
      }
      wait_async0();
#else
#pragma unroll
      for (int it = 0; it < 8; ++it) {
        const int c   = tid + it * 256;
        const int e   = c << 3;
        const int row = e >> 10, col = e & 1023;
        *(u16x8*)&Blds[row * A_STRIDE + col] = *(const u16x8*)(wsrc + e);
      }
#endif
    }
    __syncthreads();

    v8f acc = {};
#pragma unroll 4
    for (int kb = 0; kb < 32; ++kb) {
      const int ka = kb * 32 + g * 8;
      v16bf af = frag_from_lds(arow + ka, arow + ka + 16);
      v16bf bf = frag_from_lds(brow + ka, brow + ka + 16);
      acc = wmma_bf16(af, bf, acc);
    }

    // Epilogue: D element (m = v + 8g, n = lm); fold tanh + wm, reduce over n
    const float wmn = wm[n0 + lm];
#pragma unroll
    for (int v = 0; v < 8; ++v) {
      const int rl = mrow + v + 8 * g;  // local row; b = (r0+rl) % 64 = rl & 63
      const int b  = rl & 63;
      float val = wmn * tanhf(acc[v] + hidp[b * 1024 + n0 + lm]);
      val += __shfl_xor(val, 1);
      val += __shfl_xor(val, 2);
      val += __shfl_xor(val, 4);
      val += __shfl_xor(val, 8);
      sAcc[v] += val;
    }
  }

  if (lm == 0) {
#pragma unroll
    for (int v = 0; v < 8; ++v)
      scores[r0 + mrow + v + 8 * g] = sAcc[v];
  }
}

// Softmax over S (axis 0) per batch column b; applies mask.
__global__ __launch_bounds__(256) void softmax_kernel(const float* __restrict__ scores,
                                                      const float* __restrict__ mask,
                                                      float* __restrict__ alpha) {
  const int b = blockIdx.x, t = threadIdx.x;
  __shared__ float red[256];
  float v[4];
#pragma unroll
  for (int i = 0; i < 4; ++i) {
    const int s = t + i * 256;
    const float sc = scores[s * 64 + b];
    v[i] = (mask[s * 64 + b] > 0.f) ? sc : -1e8f;
  }
  float mx = fmaxf(fmaxf(v[0], v[1]), fmaxf(v[2], v[3]));
  red[t] = mx;
  __syncthreads();
  for (int o = 128; o > 0; o >>= 1) {
    if (t < o) red[t] = fmaxf(red[t], red[t + o]);
    __syncthreads();
  }
  mx = red[0];
  __syncthreads();
  float sum = 0.f;
#pragma unroll
  for (int i = 0; i < 4; ++i) { v[i] = __expf(v[i] - mx); sum += v[i]; }
  red[t] = sum;
  __syncthreads();
  for (int o = 128; o > 0; o >>= 1) {
    if (t < o) red[t] += red[t + o];
    __syncthreads();
  }
  const float inv = 1.0f / red[0];
#pragma unroll
  for (int i = 0; i < 4; ++i) {
    const int s = t + i * 256;
    alpha[s * 64 + b] = v[i] * inv;
  }
}

// z[b,c] = sum_s alpha[s,b] * ctx[s,b,c]  (memory-bound sweep over ctx)
__global__ __launch_bounds__(256) void attend_kernel(const float* __restrict__ alpha,
                                                     const float* __restrict__ ctx,
                                                     float* __restrict__ z) {
  const int b = blockIdx.y;
  const int c = blockIdx.x * 256 + threadIdx.x;
  __shared__ float al[1024];
  for (int i = threadIdx.x; i < 1024; i += 256) al[i] = alpha[i * 64 + b];
  __syncthreads();
  float acc = 0.f;
#pragma unroll 4
  for (int s = 0; s < 1024; ++s)
    acc += al[s] * ctx[(size_t)(s * 64 + b) * 1024 + c];
  z[b * 1024 + c] = acc;
}

extern "C" void kernel_launch(void* const* d_in, const int* in_sizes, int n_in,
                              void* d_out, int out_size, void* d_ws, size_t ws_size,
                              hipStream_t stream) {
  const float* hid  = (const float*)d_in[0];  // [1,64,1024]
  const float* ctx  = (const float*)d_in[1];  // [1024,64,1024]
  const float* mask = (const float*)d_in[2];  // [1024,64]
  const float* Wc   = (const float*)d_in[3];  // [1024,1024]
  const float* Wh   = (const float*)d_in[4];  // [1024,1024]
  const float* wm   = (const float*)d_in[5];  // [1,1024]
  const float* Wo   = (const float*)d_in[6];  // [1024,1024]

  float* out   = (float*)d_out;
  float* alpha = out;                   // [S,B] = 65536
  float* zt    = out + S_DIM * B_DIM;   // [B,H] = 65536

  float* ws              = (float*)d_ws;
  float* hid_ws          = ws;                   // 64*1024 fp32
  float* scores_ws       = ws + 64 * 1024;       // 1024*64 fp32
  float* z_ws            = ws + 2 * 64 * 1024;   // 64*1024 fp32
  unsigned short* Wc_bf  = (unsigned short*)(ws + 3 * 64 * 1024);  // 1024*1024 bf16

  // 0) one-shot Wc fp32 -> bf16 (2MB, stays hot in L2)
  cvt_bf16_kernel<<<1024, 256, 0, stream>>>(Wc, Wc_bf);
  // 1) hid_ = hid @ Wh^T   [64,1024]
  gemm64nt_kernel<<<8, 256, 0, stream>>>(hid, Wh, hid_ws);
  // 2) fused big GEMM + tanh + wm-dot -> raw scores [S*B]
  scores_kernel<<<512, 256, 0, stream>>>(ctx, Wc_bf, hid_ws, wm, scores_ws);
  // 3) masked softmax over S -> alpha (first output)
  softmax_kernel<<<64, 256, 0, stream>>>(scores_ws, mask, alpha);
  // 4) z = sum_s alpha * ctx   [64,1024]
  attend_kernel<<<dim3(4, 64), 256, 0, stream>>>(alpha, ctx, z_ws);
  // 5) z_t = z @ Wo^T -> second output [64,1024]
  gemm64nt_kernel<<<8, 256, 0, stream>>>(z_ws, Wo, zt);
}
